// MultiHeadAttention_584115552784
// MI455X (gfx1250) — matrix operand order repaired and checked by reading, not compiled, not run on hardware
//
#include <hip/hip_runtime.h>

typedef unsigned short u16;
typedef __attribute__((ext_vector_type(16))) __bf16 v16bf;
typedef __attribute__((ext_vector_type(8)))  float  v8f;
typedef __attribute__((ext_vector_type(4)))  int    vint4;

#if defined(__HIP_DEVICE_COMPILE__) && __has_builtin(__builtin_amdgcn_global_load_async_to_lds_b128)
#define HAVE_ASYNC_LDS 1
typedef __attribute__((address_space(1))) vint4* gbl_v4p;
typedef __attribute__((address_space(3))) vint4* lds_v4p;
__device__ __forceinline__ void async_b128(const void* g, void* l) {
  __builtin_amdgcn_global_load_async_to_lds_b128(
      (gbl_v4p)(vint4*)g, (lds_v4p)(vint4*)l, 0, 0);
}
__device__ __forceinline__ void async_wait0() {
#if __has_builtin(__builtin_amdgcn_s_wait_asynccnt)
  __builtin_amdgcn_s_wait_asynccnt(0);
#else
  asm volatile("s_wait_asynccnt 0" ::: "memory");
#endif
}
#else
#define HAVE_ASYNC_LDS 0
__device__ __forceinline__ void async_wait0() {}
#endif

union Frag {
  v16bf v;
  uint4 q[2];
  u16 s[16];
};

__device__ __forceinline__ u16 f32_to_bf16(float f) {
  union { float f; unsigned int u; } c; c.f = f;
  unsigned int u = c.u;
  unsigned int r = u + 0x7FFFu + ((u >> 16) & 1u);
  return (u16)(r >> 16);
}

__device__ __forceinline__ v8f vzero8() {
  v8f z;
#pragma unroll
  for (int i = 0; i < 8; ++i) z[i] = 0.f;
  return z;
}

// ---------------------------------------------------------------------------
// Wt[k][n] = W[n][k], f32 -> bf16
// ---------------------------------------------------------------------------
__global__ __launch_bounds__(256) void transpose_bf16_kernel(
    const float* __restrict__ W, u16* __restrict__ Wt, int D) {
  __shared__ float tile[32][33];
  const int tx = threadIdx.x;
  for (int i = threadIdx.y; i < 32; i += 8)
    tile[i][tx] = W[(size_t)(blockIdx.y * 32 + i) * D + blockIdx.x * 32 + tx];
  __syncthreads();
  for (int i = threadIdx.y; i < 32; i += 8)
    Wt[(size_t)(blockIdx.x * 32 + i) * D + blockIdx.y * 32 + tx] =
        f32_to_bf16(tile[tx][i]);
}

// ---------------------------------------------------------------------------
// C[M,N] = A[M,K] * Bt[K,N](bf16) + bias.  128x128 tile, K-step 32,
// double-buffered LDS.  A_F32: A is f32, converted during staging (fuses the
// downconvert pass).  Otherwise A is bf16 and staged via async-to-LDS.
// ---------------------------------------------------------------------------
template <bool OUT_F32, bool A_F32>
__global__ __launch_bounds__(256) void gemm_bias_kernel(
    const void* __restrict__ Ain, const u16* __restrict__ Bt,
    const float* __restrict__ bias, void* __restrict__ Cout,
    int M, int N, int K) {
  __shared__ u16 As[2][128 * 40];  // [row][k], 80B rows (16B aligned)
  __shared__ u16 Bs[2][128 * 40];  // transposed [n][k]

  const float* A32 = (const float*)Ain;
  const u16* A16 = (const u16*)Ain;

  const int tid = threadIdx.x;
  const int lane = tid & 31;
  const int wave = tid >> 5;
  const int hw = lane >> 4;
  const int l16 = lane & 15;
  const int n0 = blockIdx.x * 128;
  const int m0 = blockIdx.y * 128;
  const int rowBase = (wave & 3) * 32;
  const int colBase = (wave >> 2) * 64;

  const int aRow = tid >> 1, aCol = (tid & 1) * 16;
  const int bK = tid >> 3, bN = (tid & 7) * 16;

  // uniform bases + 32-bit per-lane offsets (saddr+voffset addressing)
  const float* gA32 = A32 + (size_t)m0 * K;
  const u16* gA16 = A16 + (size_t)m0 * K;
  const u16* gBt = Bt + n0;
  const unsigned aOff = (unsigned)(aRow * K + aCol);
  const unsigned bOff = (unsigned)(bK * N + bN);

  v8f acc[2][4];
#pragma unroll
  for (int m = 0; m < 2; ++m)
#pragma unroll
    for (int n = 0; n < 4; ++n) acc[m][n] = vzero8();

  // ---- prologue: stage K-step 0 into buffer 0 ----
  {
    if constexpr (A_F32) {
      union { uint4 q[4]; float f[16]; } aa;
      const float* p = gA32 + aOff;
#pragma unroll
      for (int i = 0; i < 4; ++i) aa.q[i] = *(const uint4*)(p + 4 * i);
      union { uint4 q[2]; u16 s[16]; } ac;
#pragma unroll
      for (int i = 0; i < 16; ++i) ac.s[i] = f32_to_bf16(aa.f[i]);
      *(uint4*)&As[0][aRow * 40 + aCol] = ac.q[0];
      *(uint4*)&As[0][aRow * 40 + aCol + 8] = ac.q[1];
    } else {
#if HAVE_ASYNC_LDS
      async_b128(gA16 + aOff, &As[0][aRow * 40 + aCol]);
      async_b128(gA16 + aOff + 8, &As[0][aRow * 40 + aCol + 8]);
#else
      uint4 a0 = *(const uint4*)(gA16 + aOff);
      uint4 a1 = *(const uint4*)(gA16 + aOff + 8);
      *(uint4*)&As[0][aRow * 40 + aCol] = a0;
      *(uint4*)&As[0][aRow * 40 + aCol + 8] = a1;
#endif
    }
    union { uint4 q[2]; u16 s[16]; } bb;
    bb.q[0] = *(const uint4*)(gBt + bOff);
    bb.q[1] = *(const uint4*)(gBt + bOff + 8);
#pragma unroll
    for (int i = 0; i < 16; ++i) Bs[0][(bN + i) * 40 + bK] = bb.s[i];
    if constexpr (!A_F32) async_wait0();
    __syncthreads();
  }

  int cur = 0;
  for (int kk = 0; kk < K; kk += 32) {
    const int nxt = cur ^ 1;
    const bool hasNext = (kk + 32) < K;

    // ---- kick off next tile's loads while we compute this one ----
    union { uint4 q[4]; float f[16]; } aaN;
    union { uint4 q[2]; u16 s[16]; } bbN;
    if (hasNext) {
      if constexpr (A_F32) {
        const float* p = gA32 + aOff + kk + 32;
#pragma unroll
        for (int i = 0; i < 4; ++i) aaN.q[i] = *(const uint4*)(p + 4 * i);
      } else {
#if HAVE_ASYNC_LDS
        async_b128(gA16 + aOff + kk + 32, &As[nxt][aRow * 40 + aCol]);
        async_b128(gA16 + aOff + kk + 32 + 8, &As[nxt][aRow * 40 + aCol + 8]);
#else
        aaN.q[0] = *(const uint4*)(gA16 + aOff + kk + 32);
        aaN.q[1] = *(const uint4*)(gA16 + aOff + kk + 32 + 8);
#endif
      }
      bbN.q[0] = *(const uint4*)(gBt + bOff + (unsigned)(kk + 32) * N);
      bbN.q[1] = *(const uint4*)(gBt + bOff + (unsigned)(kk + 32) * N + 8);
      if (kk + 64 < K) {
        __builtin_prefetch(gBt + bOff + (unsigned)(kk + 64) * N, 0, 0);
      }
    }

    // ---- fragments (two 16B contiguous halves each -> ds_load_b128) ----
    Frag af[2], bf4[4];
#pragma unroll
    for (int m = 0; m < 2; ++m) {
      const u16* ap = &As[cur][(rowBase + m * 16 + l16) * 40];
      af[m].q[0] = *(const uint4*)&ap[hw * 8];
      af[m].q[1] = *(const uint4*)&ap[16 + hw * 8];
    }
#pragma unroll
    for (int n = 0; n < 4; ++n) {
      const u16* bp = &Bs[cur][(colBase + n * 16 + l16) * 40];
      bf4[n].q[0] = *(const uint4*)&bp[hw * 8];
      bf4[n].q[1] = *(const uint4*)&bp[16 + hw * 8];
    }

#pragma unroll
    for (int m = 0; m < 2; ++m)
#pragma unroll
      for (int n = 0; n < 4; ++n)
        acc[m][n] = __builtin_amdgcn_wmma_f32_16x16x32_bf16(
            false, af[m].v, false, bf4[n].v, (short)0, acc[m][n], false, false);

    // ---- drain next tile into the alternate buffer ----
    if (hasNext) {
      if constexpr (A_F32) {
        union { uint4 q[2]; u16 s[16]; } ac;
#pragma unroll
        for (int i = 0; i < 16; ++i) ac.s[i] = f32_to_bf16(aaN.f[i]);
        *(uint4*)&As[nxt][aRow * 40 + aCol] = ac.q[0];
        *(uint4*)&As[nxt][aRow * 40 + aCol + 8] = ac.q[1];
      } else {
#if !HAVE_ASYNC_LDS
        *(uint4*)&As[nxt][aRow * 40 + aCol] = aaN.q[0];
        *(uint4*)&As[nxt][aRow * 40 + aCol + 8] = aaN.q[1];
#endif
      }
#pragma unroll
      for (int i = 0; i < 16; ++i) Bs[nxt][(bN + i) * 40 + bK] = bbN.s[i];
    }
    if constexpr (!A_F32) async_wait0();
    __syncthreads();
    cur = nxt;
  }

#pragma unroll
  for (int m = 0; m < 2; ++m) {
#pragma unroll
    for (int n = 0; n < 4; ++n) {
      const int gCol = n0 + colBase + n * 16 + l16;
      const float bv = bias[gCol];
#pragma unroll
      for (int r = 0; r < 8; ++r) {
        const int gRow = m0 + rowBase + m * 16 + hw * 8 + r;
        const float v = acc[m][n][r] + bv;
        if constexpr (OUT_F32)
          ((float*)Cout)[(size_t)gRow * N + gCol] = v;
        else
          ((u16*)Cout)[(size_t)gRow * N + gCol] = f32_to_bf16(v);
      }
    }
  }
}

// ---------------------------------------------------------------------------
// Flash attention: 4 waves/block, 16 queries per wave, shared K/V staging.
// ---------------------------------------------------------------------------
__global__ __launch_bounds__(128) void attn_kernel(
    const u16* __restrict__ Q, const u16* __restrict__ Kh,
    const u16* __restrict__ V, const unsigned char* __restrict__ msk,
    u16* __restrict__ Out, int S, int D) {
  const int tid = threadIdx.x;
  const int wave = tid >> 5;
  const int lane = tid & 31;
  const int hw = lane >> 4, l16 = lane & 15;
  const int qb = blockIdx.x * 64 + wave * 16;
  const int h = blockIdx.y, b = blockIdx.z;
  const size_t rb = (size_t)b * S;
  const int cOff = h * 64;

  __shared__ u16 Ks[32 * 72];               // [key][dim64+pad]  (144B rows)
  __shared__ u16 Vt[64 * 40];               // [dim][key32+pad]  (80B rows)
  __shared__ unsigned char Ml[4][16 * 32];  // per-wave mask tile
  __shared__ u16 Pl[4][16 * 40];            // per-wave P tile (80B rows)

  const int sKey = tid >> 2;        // 0..31 staging key
  const int sSeg = (tid & 3) * 16;  // staging dim segment

  // uniform bases + 32-bit per-lane offsets
  const u16* Qb = Q + rb * D + cOff;
  const u16* Kb = Kh + rb * D + cOff;
  const u16* Vb = V + rb * D + cOff;
  u16* Ob = Out + rb * D + cOff;
  const unsigned char* mrow = msk + ((size_t)b * S + qb) * S;

  // Q fragments (A-layout): two 16B contiguous halves per K32 fragment
  Frag qa[2];
  const unsigned qoff = (unsigned)((qb + l16) * D);
  qa[0].q[0] = *(const uint4*)(Qb + qoff + hw * 8);
  qa[0].q[1] = *(const uint4*)(Qb + qoff + 16 + hw * 8);
  qa[1].q[0] = *(const uint4*)(Qb + qoff + 32 + hw * 8);
  qa[1].q[1] = *(const uint4*)(Qb + qoff + 48 + hw * 8);

  v8f o[4];
#pragma unroll
  for (int dg = 0; dg < 4; ++dg) o[dg] = vzero8();
  float mrun[8], lrun[8];
#pragma unroll
  for (int r = 0; r < 8; ++r) { mrun[r] = -1e30f; lrun[r] = 0.f; }

  for (int kb = 0; kb < S; kb += 32) {
    const unsigned svOff = (unsigned)((kb + sKey) * D + sSeg);
    // ---- stage K tile (row-major, contiguous) ----
#if HAVE_ASYNC_LDS
    async_b128(Kb + svOff, &Ks[sKey * 72 + sSeg]);
    async_b128(Kb + svOff + 8, &Ks[sKey * 72 + sSeg + 8]);
#else
    uint4 ka = *(const uint4*)(Kb + svOff);
    uint4 kc = *(const uint4*)(Kb + svOff + 8);
    *(uint4*)&Ks[sKey * 72 + sSeg] = ka;
    *(uint4*)&Ks[sKey * 72 + sSeg + 8] = kc;
#endif
    // ---- stage V tile transposed: Vt[dim][key] ----
    union { uint4 q[2]; u16 s[16]; } vv;
    vv.q[0] = *(const uint4*)(Vb + svOff);
    vv.q[1] = *(const uint4*)(Vb + svOff + 8);
#pragma unroll
    for (int i = 0; i < 16; ++i) Vt[(sSeg + i) * 40 + sKey] = vv.s[i];

    // ---- stage this wave's mask tile + all-true fast-path vote ----
    uint4 mq = *(const uint4*)(mrow + (unsigned)(l16 * S + kb + hw * 16));
    *(uint4*)&Ml[wave][l16 * 32 + hw * 16] = mq;
    const bool allset =
        __all((mq.x & mq.y & mq.z & mq.w) == 0x01010101u);

    if (kb + 32 < S) {
      __builtin_prefetch(Kb + svOff + 32 * D, 0, 0);
      __builtin_prefetch(Vb + svOff + 32 * D, 0, 0);
    }

#if HAVE_ASYNC_LDS
    async_wait0();
#endif
    __syncthreads();

    // ---- scores: two 16x16 tiles, each 2 WMMAs over DS=64 ----
    float ps[2][8];
#pragma unroll
    for (int t = 0; t < 2; ++t) {
      const u16* kk = &Ks[(t * 16 + l16) * 72];  // lane -> key
      Frag k0, k1;                               // element e -> dim
      k0.q[0] = *(const uint4*)&kk[hw * 8];
      k0.q[1] = *(const uint4*)&kk[16 + hw * 8];
      k1.q[0] = *(const uint4*)&kk[32 + hw * 8];
      k1.q[1] = *(const uint4*)&kk[48 + hw * 8];
      v8f st = vzero8();
      st = __builtin_amdgcn_wmma_f32_16x16x32_bf16(false, qa[0].v, false, k0.v,
                                                   (short)0, st, false, false);
      st = __builtin_amdgcn_wmma_f32_16x16x32_bf16(false, qa[1].v, false, k1.v,
                                                   (short)0, st, false, false);
#pragma unroll
      for (int r = 0; r < 8; ++r) ps[t][r] = st[r] * 0.125f;  // 1/sqrt(64)
    }
    if (!allset) {  // rare wave-uniform slow path: apply per-element mask
#pragma unroll
      for (int t = 0; t < 2; ++t)
#pragma unroll
        for (int r = 0; r < 8; ++r)
          if (!Ml[wave][(hw * 8 + r) * 32 + t * 16 + l16]) ps[t][r] = -1e9f;
    }

    // ---- streaming softmax (rows live across 16 lanes of each half) ----
#pragma unroll
    for (int r = 0; r < 8; ++r) {
      float tm = fmaxf(ps[0][r], ps[1][r]);
      tm = fmaxf(tm, __shfl_xor(tm, 1, 32));
      tm = fmaxf(tm, __shfl_xor(tm, 2, 32));
      tm = fmaxf(tm, __shfl_xor(tm, 4, 32));
      tm = fmaxf(tm, __shfl_xor(tm, 8, 32));
      const float mnew = fmaxf(mrun[r], tm);
      const float alpha = __expf(mrun[r] - mnew);
      const float p0 = __expf(ps[0][r] - mnew);
      const float p1 = __expf(ps[1][r] - mnew);
      float rs = p0 + p1;
      rs += __shfl_xor(rs, 1, 32);
      rs += __shfl_xor(rs, 2, 32);
      rs += __shfl_xor(rs, 4, 32);
      rs += __shfl_xor(rs, 8, 32);
      lrun[r] = lrun[r] * alpha + rs;
      mrun[r] = mnew;
#pragma unroll
      for (int dg = 0; dg < 4; ++dg) o[dg][r] *= alpha;
      const int row = hw * 8 + r;  // C-layout row
      Pl[wave][row * 40 + l16] = f32_to_bf16(p0);
      Pl[wave][row * 40 + 16 + l16] = f32_to_bf16(p1);
    }
    __builtin_amdgcn_wave_barrier();  // per-wave DS ops are in-order (ISA 7.3)

    // ---- P (A-layout) and P*V ----
    Frag pa;
    pa.q[0] = *(const uint4*)&Pl[wave][l16 * 40 + hw * 8];
    pa.q[1] = *(const uint4*)&Pl[wave][l16 * 40 + 16 + hw * 8];
#pragma unroll
    for (int dg = 0; dg < 4; ++dg) {
      const u16* vp = &Vt[(dg * 16 + l16) * 40];  // lane -> dim
      Frag vf;                                    // element e -> key
      vf.q[0] = *(const uint4*)&vp[hw * 8];
      vf.q[1] = *(const uint4*)&vp[16 + hw * 8];
      o[dg] = __builtin_amdgcn_wmma_f32_16x16x32_bf16(
          false, pa.v, false, vf.v, (short)0, o[dg], false, false);
    }
    __syncthreads();  // protect Ks/Vt before next staging round
  }

#pragma unroll
  for (int dg = 0; dg < 4; ++dg) {
#pragma unroll
    for (int r = 0; r < 8; ++r) {
      const unsigned ooff = (unsigned)((qb + hw * 8 + r) * D + dg * 16 + l16);
      Ob[ooff] = f32_to_bf16(o[dg][r] / lrun[r]);
    }
  }
}

// ---------------------------------------------------------------------------
extern "C" void kernel_launch(void* const* d_in, const int* in_sizes, int n_in,
                              void* d_out, int out_size, void* d_ws,
                              size_t ws_size, hipStream_t stream) {
  constexpr int B = 2, S = 2048, D = 1024, H = 16;
  constexpr int M = B * S;

  const float* q = (const float*)d_in[0];
  const float* k = (const float*)d_in[1];
  const float* v = (const float*)d_in[2];
  const unsigned char* mask = (const unsigned char*)d_in[3];
  const float* Wq = (const float*)d_in[4];
  const float* bq = (const float*)d_in[5];
  const float* Wk = (const float*)d_in[6];
  const float* bk = (const float*)d_in[7];
  const float* Wv = (const float*)d_in[8];
  const float* bv = (const float*)d_in[9];
  const float* Wo = (const float*)d_in[10];
  const float* bo = (const float*)d_in[11];
  float* out = (float*)d_out;

  char* ws = (char*)d_ws;
  size_t off = 0;
  auto take = [&](size_t bytes) -> void* {
    void* p = ws + off;
    off += (bytes + 255) & ~(size_t)255;
    return p;
  };
  const size_t nTok = (size_t)M * D;
  u16* WqT = (u16*)take((size_t)D * D * 2);
  u16* WkT = (u16*)take((size_t)D * D * 2);
  u16* WvT = (u16*)take((size_t)D * D * 2);
  u16* WoT = (u16*)take((size_t)D * D * 2);
  u16* Qp = (u16*)take(nTok * 2);
  u16* Kp = (u16*)take(nTok * 2);
  u16* Vp = (u16*)take(nTok * 2);
  u16* Cc = (u16*)take(nTok * 2);

  // weights -> bf16 W^T
  dim3 tg(D / 32, D / 32), tb(32, 8);
  transpose_bf16_kernel<<<tg, tb, 0, stream>>>(Wq, WqT, D);
  transpose_bf16_kernel<<<tg, tb, 0, stream>>>(Wk, WkT, D);
  transpose_bf16_kernel<<<tg, tb, 0, stream>>>(Wv, WvT, D);
  transpose_bf16_kernel<<<tg, tb, 0, stream>>>(Wo, WoT, D);

  // Q/K/V projections: f32 activations converted in-flight, bias fused
  dim3 gg(D / 128, M / 128);
  gemm_bias_kernel<false, true><<<gg, 256, 0, stream>>>(q, WqT, bq, Qp, M, D, D);
  gemm_bias_kernel<false, true><<<gg, 256, 0, stream>>>(k, WkT, bk, Kp, M, D, D);
  gemm_bias_kernel<false, true><<<gg, 256, 0, stream>>>(v, WvT, bv, Vp, M, D, D);

  // flash attention
  attn_kernel<<<dim3(S / 64, H, B), 128, 0, stream>>>(Qp, Kp, Vp, mask, Cc, S, D);

  // output projection, f32 + bias straight into d_out (bf16 A via async LDS)
  gemm_bias_kernel<true, false><<<gg, 256, 0, stream>>>(Cc, WoT, bo, out, M, D, D);
}
